// GACFV2_layer_41051297415860
// MI455X (gfx1250) — matrix operand order repaired and checked by reading, not compiled, not run on hardware
//
#include <hip/hip_runtime.h>
#include <hip/hip_bf16.h>

// ---------------------------------------------------------------------------
// GACF-V2 forward for MI455X (gfx1250, wave32, WMMA bf16)
// N=2625 nodes padded to NP=2688 rows; all GEMMs via v_wmma_f32_16x16x32_bf16
// ---------------------------------------------------------------------------

#define NNODE 2625
#define NPAD  2688          // 21 * 128
#define OUTW  448

typedef __attribute__((ext_vector_type(16))) __bf16          v16bf;
typedef __attribute__((ext_vector_type(16))) unsigned short  us16;
typedef __attribute__((ext_vector_type(8)))  unsigned short  us8;
typedef __attribute__((ext_vector_type(4)))  unsigned short  us4;
typedef __attribute__((ext_vector_type(8)))  float           v8f;

static __device__ __forceinline__ unsigned short f2bf(float f) {
    unsigned int u = __builtin_bit_cast(unsigned int, f);
    u += 0x7FFFu + ((u >> 16) & 1u);          // round-to-nearest-even
    return (unsigned short)(u >> 16);
}

static __device__ __forceinline__ v8f wmma_bf16(us16 a, us16 b, v8f c) {
    return __builtin_amdgcn_wmma_f32_16x16x32_bf16(
        false, __builtin_bit_cast(v16bf, a),
        false, __builtin_bit_cast(v16bf, b),
        (short)0, c, false, false);
}

// ---------------------------------------------------------------------------
// init: features = concat(uEmbd, iEmbd), zero pad rows, write out cols [0,256)
// ---------------------------------------------------------------------------
__global__ __launch_bounds__(256) void k_init(const float* __restrict__ uE,
                                              const float* __restrict__ iE,
                                              float* __restrict__ X,
                                              float* __restrict__ out) {
    int i = blockIdx.x * 256 + threadIdx.x;          // over NPAD*256
    int r = i >> 8, c = i & 255;
    float v = 0.0f;
    if (r < 943)            v = uE[r * 256 + c];
    else if (r < NNODE)     v = iE[(r - 943) * 256 + c];
    X[i] = v;
    if (r < NNODE) out[(size_t)r * OUTW + c] = v;
}

// ---------------------------------------------------------------------------
// GEMM: C[NPAD,Nc] = [relu](A[NPAD,K] @ W[K,Nc] + bias)   (K mult 32, Nc mult 64)
// grid(NPAD/128, Nc/64), block 256 (8 waves). 128x64 tile, 4 WMMA / wave / step.
// B-fragment loads are hoisted ahead of the WMMA group so the scheduler can
// overlap ds_load latency with earlier WMMAs (partial s_wait_dscnt).
// ---------------------------------------------------------------------------
__global__ __launch_bounds__(256) void k_gemm(const float* __restrict__ A,
                                              const float* __restrict__ W,
                                              const float* __restrict__ bias,
                                              float* __restrict__ C,
                                              int K, int Nc, int relu) {
    __shared__ __align__(16) unsigned short lsA[128 * 32];
    __shared__ __align__(32) unsigned short lsB[64 * 32];   // transposed: [n][k]

    const int t    = threadIdx.x;
    const int m0   = blockIdx.x * 128;
    const int n0   = blockIdx.y * 64;
    const int w    = t >> 5;
    const int lane = t & 31;
    const int nn   = lane & 15;
    const int half = lane >> 4;

    v8f acc[4];
#pragma unroll
    for (int ct = 0; ct < 4; ++ct)
        acc[ct] = (v8f){0.f,0.f,0.f,0.f,0.f,0.f,0.f,0.f};

    const int ar = t >> 1, seg = t & 1;      // A staging: 128 rows x 2 segs
    const int bk = t & 31, ng  = t >> 5;     // B staging: 32 k x 8 col-groups of 8

    for (int k0 = 0; k0 < K; k0 += 32) {
        { // stage A tile (128x32) as bf16
            const float* ap = &A[(size_t)(m0 + ar) * K + k0 + seg * 16];
            __builtin_prefetch(ap + 32, 0, 1);
            const float4 f0 = ((const float4*)ap)[0];
            const float4 f1 = ((const float4*)ap)[1];
            const float4 f2 = ((const float4*)ap)[2];
            const float4 f3 = ((const float4*)ap)[3];
            us8 p0, p1;
            p0[0]=f2bf(f0.x); p0[1]=f2bf(f0.y); p0[2]=f2bf(f0.z); p0[3]=f2bf(f0.w);
            p0[4]=f2bf(f1.x); p0[5]=f2bf(f1.y); p0[6]=f2bf(f1.z); p0[7]=f2bf(f1.w);
            p1[0]=f2bf(f2.x); p1[1]=f2bf(f2.y); p1[2]=f2bf(f2.z); p1[3]=f2bf(f2.w);
            p1[4]=f2bf(f3.x); p1[5]=f2bf(f3.y); p1[6]=f2bf(f3.z); p1[7]=f2bf(f3.w);
            *(us8*)&lsA[ar * 32 + seg * 16]     = p0;
            *(us8*)&lsA[ar * 32 + seg * 16 + 8] = p1;
        }
        { // stage B tile (32x64) transposed as bf16
            const float* bp = &W[(size_t)(k0 + bk) * Nc + n0 + ng * 8];
            __builtin_prefetch(bp + 32 * Nc, 0, 1);
            const float4 g0 = ((const float4*)bp)[0];
            const float4 g1 = ((const float4*)bp)[1];
            lsB[(ng * 8 + 0) * 32 + bk] = f2bf(g0.x);
            lsB[(ng * 8 + 1) * 32 + bk] = f2bf(g0.y);
            lsB[(ng * 8 + 2) * 32 + bk] = f2bf(g0.z);
            lsB[(ng * 8 + 3) * 32 + bk] = f2bf(g0.w);
            lsB[(ng * 8 + 4) * 32 + bk] = f2bf(g1.x);
            lsB[(ng * 8 + 5) * 32 + bk] = f2bf(g1.y);
            lsB[(ng * 8 + 6) * 32 + bk] = f2bf(g1.z);
            lsB[(ng * 8 + 7) * 32 + bk] = f2bf(g1.w);
        }
        __syncthreads();

        const us8 alo = *(const us8*)&lsA[(16 * w + nn) * 32 + 16 * half];
        const us8 ahi = *(const us8*)&lsA[(16 * w + nn) * 32 + 16 * half + 8];
        const us16 af = __builtin_shufflevector(alo, ahi,
                            0,1,2,3,4,5,6,7,8,9,10,11,12,13,14,15);

        us16 bfr[4];
#pragma unroll
        for (int ct = 0; ct < 4; ++ct)
            bfr[ct] = *(const us16*)&lsB[(nn + 16 * ct) * 32 + 16 * half];
#pragma unroll
        for (int ct = 0; ct < 4; ++ct)
            acc[ct] = wmma_bf16(af, bfr[ct], acc[ct]);

        __syncthreads();
    }

#pragma unroll
    for (int ct = 0; ct < 4; ++ct) {
        const float bb = bias[n0 + 16 * ct + nn];
#pragma unroll
        for (int r = 0; r < 8; ++r) {
            const int row = m0 + 16 * w + r + 8 * half;
            float v = acc[ct][r] + bb;
            if (relu) v = fmaxf(v, 0.f);
            C[(size_t)row * Nc + n0 + 16 * ct + nn] = v;
        }
    }
}

// ---------------------------------------------------------------------------
// f32 -> bf16 (same layout), vectorized x4
// ---------------------------------------------------------------------------
__global__ __launch_bounds__(256) void k_tobf16(const float* __restrict__ X,
                                                unsigned short* __restrict__ Y) {
    const int i = (blockIdx.x * 256 + threadIdx.x) * 4;
    const float4 f = *(const float4*)&X[i];
    us4 p;
    p[0] = f2bf(f.x); p[1] = f2bf(f.y); p[2] = f2bf(f.z); p[3] = f2bf(f.w);
    *(us4*)&Y[i] = p;
}

// ---------------------------------------------------------------------------
// f32 [NPAD,F] -> bf16 transposed [F,NPAD]  (output-index contiguous)
// ---------------------------------------------------------------------------
__global__ __launch_bounds__(256) void k_tr_bf16(const float* __restrict__ X,
                                                 unsigned short* __restrict__ Y, int F) {
    const int i = blockIdx.x * 256 + threadIdx.x;    // over F*NPAD
    const int c = i / NPAD, r = i - c * NPAD;
    Y[i] = f2bf(X[(size_t)r * F + c]);
}

// ---------------------------------------------------------------------------
// Flash-style masked MHA. grid(NPAD/16, HEADS), block 32 (1 wave).
// Qb,Kb: bf16 [NPAD,F]; Vt: bf16 [F,NPAD]; O: f32 [NPAD,F]. dk in {16,32}.
// ---------------------------------------------------------------------------
__global__ __launch_bounds__(32) void k_attn(const unsigned short* __restrict__ Qb,
                                             const unsigned short* __restrict__ Kb,
                                             const unsigned short* __restrict__ Vt,
                                             float* __restrict__ O,
                                             const unsigned char* __restrict__ mask,
                                             int F, int dk, float scale) {
    __shared__ __align__(16) unsigned short lp[16 * 32];   // P tile bf16

    const int m0   = blockIdx.x * 16;
    const int h    = blockIdx.y;
    const int hoff = h * dk;
    const int lane = threadIdx.x & 31;
    const int nn   = lane & 15;
    const int half = lane >> 4;

    const us16 zero16 = {0,0,0,0,0,0,0,0,0,0,0,0,0,0,0,0};

    // ---- Q fragment (A layout), dk padded to 32 with zeros
    us16 qa;
    {
        const us8 qlo = *(const us8*)&Qb[(size_t)(m0 + nn) * F + hoff + 16 * half];
        us8 qhi = {0,0,0,0,0,0,0,0};
        if (dk == 32 || half == 0)
            qhi = *(const us8*)&Qb[(size_t)(m0 + nn) * F + hoff + 16 * half + 8];
        qa = __builtin_shufflevector(qlo, qhi,
                 0,1,2,3,4,5,6,7,8,9,10,11,12,13,14,15);
    }

    v8f o0 = {0.f,0.f,0.f,0.f,0.f,0.f,0.f,0.f};
    v8f o1 = {0.f,0.f,0.f,0.f,0.f,0.f,0.f,0.f};
    float mrow[8], lrow[8];
#pragma unroll
    for (int r = 0; r < 8; ++r) { mrow[r] = -1e38f; lrow[r] = 0.f; }

    for (int j0 = 0; j0 < NNODE; j0 += 32) {
        // ---- K fragments (B layout): element e <-> kk = 16*half + e
        us16 kf0 = zero16, kf1 = zero16;
        if ((16 * half) < dk) {
            kf0 = *(const us16*)&Kb[(size_t)(j0 + nn) * F + hoff + 16 * half];
            kf1 = *(const us16*)&Kb[(size_t)(j0 + 16 + nn) * F + hoff + 16 * half];
        }
        v8f s0 = {0.f,0.f,0.f,0.f,0.f,0.f,0.f,0.f};
        v8f s1 = {0.f,0.f,0.f,0.f,0.f,0.f,0.f,0.f};
        s0 = wmma_bf16(qa, kf0, s0);
        s1 = wmma_bf16(qa, kf1, s1);

        // ---- scale + mask + online softmax (rows r+8*half, cols j0+nn[+16])
        const int c0i = j0 + nn;
        const int c1i = c0i + 16;
#pragma unroll
        for (int r = 0; r < 8; ++r) {
            const int row = m0 + r + 8 * half;
            const int mr  = row < NNODE ? row : (NNODE - 1);
            float a0 = (c0i >= NNODE) ? -1e30f
                     : (mask[(size_t)mr * NNODE + c0i] ? s0[r] * scale : -1e9f);
            float a1 = (c1i >= NNODE) ? -1e30f
                     : (mask[(size_t)mr * NNODE + c1i] ? s1[r] * scale : -1e9f);
            float vm = fmaxf(a0, a1);
            vm = fmaxf(vm, __shfl_xor(vm, 1));
            vm = fmaxf(vm, __shfl_xor(vm, 2));
            vm = fmaxf(vm, __shfl_xor(vm, 4));
            vm = fmaxf(vm, __shfl_xor(vm, 8));
            const float mnew  = fmaxf(mrow[r], vm);
            const float alpha = __expf(mrow[r] - mnew);
            const float p0 = __expf(a0 - mnew);
            const float p1 = __expf(a1 - mnew);
            float rs = p0 + p1;
            rs += __shfl_xor(rs, 1);
            rs += __shfl_xor(rs, 2);
            rs += __shfl_xor(rs, 4);
            rs += __shfl_xor(rs, 8);
            lrow[r] = lrow[r] * alpha + rs;
            mrow[r] = mnew;
            o0[r] *= alpha;
            o1[r] *= alpha;
            lp[(r + 8 * half) * 32 + nn]      = f2bf(p0);
            lp[(r + 8 * half) * 32 + 16 + nn] = f2bf(p1);
        }
        asm volatile("s_wait_dscnt 0x0" ::: "memory");   // single wave: no barrier

        // ---- P fragment (A layout) from LDS
        const us8 plo = *(const us8*)&lp[nn * 32 + 16 * half];
        const us8 phi = *(const us8*)&lp[nn * 32 + 16 * half + 8];
        const us16 pa = __builtin_shufflevector(plo, phi,
                            0,1,2,3,4,5,6,7,8,9,10,11,12,13,14,15);

        // ---- V fragments (B layout) from transposed bf16 V: contiguous loads,
        //      both issued before the WMMA group
        if (dk == 32) {
            const us16 vf0 = *(const us16*)&Vt[(size_t)(hoff + nn) * NPAD + j0 + 16 * half];
            const us16 vf1 = *(const us16*)&Vt[(size_t)(hoff + 16 + nn) * NPAD + j0 + 16 * half];
            o0 = wmma_bf16(pa, vf0, o0);
            o1 = wmma_bf16(pa, vf1, o1);
        } else {
            const us16 vf0 = *(const us16*)&Vt[(size_t)(hoff + nn) * NPAD + j0 + 16 * half];
            o0 = wmma_bf16(pa, vf0, o0);
        }
    }

    // ---- finalize: O = O / l
#pragma unroll
    for (int r = 0; r < 8; ++r) {
        const int row = m0 + r + 8 * half;
        const float inv = 1.0f / lrow[r];
        O[(size_t)row * F + hoff + nn] = o0[r] * inv;
        if (dk == 32) O[(size_t)row * F + hoff + 16 + nn] = o1[r] * inv;
    }
}

// ---------------------------------------------------------------------------
// SpMM scatter: Y[row,:] += val * X[col,:]   (Y pre-initialized with X2)
// ---------------------------------------------------------------------------
__global__ void k_spmm(const int* __restrict__ rows, const int* __restrict__ cols,
                       const float* __restrict__ vals,
                       const float* __restrict__ X, float* __restrict__ Y, int F) {
    const int e = blockIdx.x;
    const int r = rows[e], c = cols[e];
    const float v = vals[e];
    atomicAdd(&Y[(size_t)r * F + threadIdx.x], v * X[(size_t)c * F + threadIdx.x]);
}

// ---------------------------------------------------------------------------
// copy slice into concatenated output
// ---------------------------------------------------------------------------
__global__ __launch_bounds__(256) void k_copy_out(const float* __restrict__ src, int W,
                                                  float* __restrict__ out, int off) {
    int i = blockIdx.x * 256 + threadIdx.x;
    if (i >= NNODE * W) return;
    int r = i / W, c = i - r * W;
    out[(size_t)r * OUTW + off + c] = src[(size_t)r * W + c];
}

// ---------------------------------------------------------------------------
extern "C" void kernel_launch(void* const* d_in, const int* in_sizes, int n_in,
                              void* d_out, int out_size, void* d_ws, size_t ws_size,
                              hipStream_t stream) {
    const unsigned char* mask = (const unsigned char*)d_in[2];
    const float* uE   = (const float*)d_in[3];
    const float* iE   = (const float*)d_in[4];
    const int*   Lidx = (const int*)d_in[5];
    const float* Lval = (const float*)d_in[6];
    const int nnz     = in_sizes[6];
    const int* Lrows  = Lidx;
    const int* Lcols  = Lidx + nnz;
    float* out = (float*)d_out;

    const size_t bufElems = (size_t)NPAD * 256;
    float* B[6];
    for (int i = 0; i < 6; ++i) B[i] = (float*)d_ws + (size_t)i * bufElems;
    unsigned short* bfb = (unsigned short*)((float*)d_ws + 6 * bufElems);
    unsigned short* Qb  = bfb;                 // bf16 [NPAD, F]
    unsigned short* Kbb = bfb + bufElems;      // bf16 [NPAD, F]
    unsigned short* Vt  = bfb + 2 * bufElems;  // bf16 [F, NPAD]

    // layer 0 params
    const float *Wq0=(const float*)d_in[7],  *Wk0=(const float*)d_in[8];
    const float *Wv0=(const float*)d_in[9],  *Wo0=(const float*)d_in[10];
    const float *bq0=(const float*)d_in[11], *bk0=(const float*)d_in[12];
    const float *bv0=(const float*)d_in[13], *bo0=(const float*)d_in[14];
    const float *aW0=(const float*)d_in[15], *ab0=(const float*)d_in[16];
    // layer 1 params
    const float *Wq1=(const float*)d_in[17], *Wk1=(const float*)d_in[18];
    const float *Wv1=(const float*)d_in[19], *Wo1=(const float*)d_in[20];
    const float *bq1=(const float*)d_in[21], *bk1=(const float*)d_in[22];
    const float *bv1=(const float*)d_in[23], *bo1=(const float*)d_in[24];
    const float *aW1=(const float*)d_in[25], *ab1=(const float*)d_in[26];

    k_init<<<(NPAD * 256) / 256, 256, 0, stream>>>(uE, iE, B[0], out);

    const dim3 gA(NPAD / 16, 8);
    const dim3 g256(NPAD / 128, 256 / 64);
    const dim3 g128(NPAD / 128, 128 / 64);
    const dim3 g64 (NPAD / 128,  64 / 64);

    // ---------------- layer 0 (F=256, dk=32, T=128) ----------------
    k_gemm<<<g256, 256, 0, stream>>>(B[0], Wq0, bq0, B[1], 256, 256, 0);
    k_gemm<<<g256, 256, 0, stream>>>(B[0], Wk0, bk0, B[2], 256, 256, 0);
    k_gemm<<<g256, 256, 0, stream>>>(B[0], Wv0, bv0, B[3], 256, 256, 0);
    k_tobf16<<<(NPAD * 256) / 1024, 256, 0, stream>>>(B[1], Qb);
    k_tobf16<<<(NPAD * 256) / 1024, 256, 0, stream>>>(B[2], Kbb);
    k_tr_bf16<<<(NPAD * 256) / 256, 256, 0, stream>>>(B[3], Vt, 256);
    k_attn<<<gA, 32, 0, stream>>>(Qb, Kbb, Vt, B[4], mask, 256, 32,
                                  0.1767766952966369f /* 1/sqrt(32) */);
    k_gemm<<<g256, 256, 0, stream>>>(B[4], Wo0, bo0, B[5], 256, 256, 0);
    hipMemcpyAsync(B[1], B[5], (size_t)NPAD * 256 * sizeof(float),
                   hipMemcpyDeviceToDevice, stream);
    k_spmm<<<nnz, 256, 0, stream>>>(Lrows, Lcols, Lval, B[5], B[1], 256);
    k_gemm<<<g128, 256, 0, stream>>>(B[1], aW0, ab0, B[2], 256, 128, 1);
    k_copy_out<<<(NNODE * 128 + 255) / 256, 256, 0, stream>>>(B[2], 128, out, 256);

    // ---------------- layer 1 (F=128, dk=16, T=64) ----------------
    k_gemm<<<g128, 256, 0, stream>>>(B[2], Wq1, bq1, B[3], 128, 128, 0);
    k_gemm<<<g128, 256, 0, stream>>>(B[2], Wk1, bk1, B[4], 128, 128, 0);
    k_gemm<<<g128, 256, 0, stream>>>(B[2], Wv1, bv1, B[5], 128, 128, 0);
    k_tobf16<<<(NPAD * 128) / 1024, 256, 0, stream>>>(B[3], Qb);
    k_tobf16<<<(NPAD * 128) / 1024, 256, 0, stream>>>(B[4], Kbb);
    k_tr_bf16<<<(NPAD * 128) / 256, 256, 0, stream>>>(B[5], Vt, 128);
    k_attn<<<gA, 32, 0, stream>>>(Qb, Kbb, Vt, B[0], mask, 128, 16,
                                  0.25f /* 1/sqrt(16) */);
    k_gemm<<<g128, 256, 0, stream>>>(B[0], Wo1, bo1, B[1], 128, 128, 0);
    hipMemcpyAsync(B[3], B[1], (size_t)NPAD * 128 * sizeof(float),
                   hipMemcpyDeviceToDevice, stream);
    k_spmm<<<nnz, 128, 0, stream>>>(Lrows, Lcols, Lval, B[1], B[3], 128);
    k_gemm<<<g64, 256, 0, stream>>>(B[3], aW1, ab1, B[4], 128, 64, 1);
    k_copy_out<<<(NNODE * 64 + 255) / 256, 256, 0, stream>>>(B[4], 64, out, 384);
}
